// AGNNConvOp_61976378081385
// MI455X (gfx1250) — matrix operand order repaired and checked by reading, compiled, not run on hardware
//
#include <hip/hip_runtime.h>
#include <hip/hip_bf16.h>
#include <cstdint>

typedef __attribute__((ext_vector_type(16))) _Float16 v16h;
typedef __attribute__((ext_vector_type(8)))  _Float16 v8h;
typedef __attribute__((ext_vector_type(4)))  _Float16 v4h;
typedef __attribute__((ext_vector_type(8)))  float    v8f;

#define DFEAT 256
#define AGNN_EPS 1e-16f

__device__ __forceinline__ float wave_max32(float v) {
#pragma unroll
  for (int o = 16; o > 0; o >>= 1) v = fmaxf(v, __shfl_xor(v, o, 32));
  return v;
}
__device__ __forceinline__ float wave_sum32(float v) {
#pragma unroll
  for (int o = 16; o > 0; o >>= 1) v += __shfl_xor(v, o, 32);
  return v;
}

// ---------------- CSR build (once per call; src/dst constant across layers) ----

__global__ void k_count(const int* __restrict__ dst, int* __restrict__ deg, int E) {
  int e = blockIdx.x * blockDim.x + threadIdx.x;
  if (e < E) atomicAdd(&deg[dst[e]], 1);
}

__global__ void k_scan(const int* __restrict__ deg, int* __restrict__ offs, int N) {
  __shared__ int part[1024];
  const int t = threadIdx.x;
  const int PER = (N + 1023) >> 10;
  const int base = t * PER;
  int s = 0;
  for (int i = 0; i < PER; ++i) {
    int idx = base + i;
    if (idx < N) s += deg[idx];
  }
  part[t] = s;
  __syncthreads();
  for (int off = 1; off < 1024; off <<= 1) {
    int v = (t >= off) ? part[t - off] : 0;
    __syncthreads();
    part[t] += v;
    __syncthreads();
  }
  int run = (t > 0) ? part[t - 1] : 0;
  for (int i = 0; i < PER; ++i) {
    int idx = base + i;
    if (idx < N) { offs[idx] = run; run += deg[idx]; }
  }
  if (t == 1023) offs[N] = part[1023];
}

__global__ void k_fill(const int* __restrict__ dst, const int* __restrict__ offs,
                       int* __restrict__ cursor, int* __restrict__ csr_eid, int E) {
  int e = blockIdx.x * blockDim.x + threadIdx.x;
  if (e < E) {
    int d = dst[e];
    int p = atomicAdd(&cursor[d], 1);
    csr_eid[offs[d] + p] = e;
  }
}

// ---------------- per-layer: row normalize, emit f16 normalized copy ----------

__global__ void k_norm(const float* __restrict__ h, _Float16* __restrict__ hn, int N) {
  int wid  = (blockIdx.x * blockDim.x + threadIdx.x) >> 5;
  int lane = threadIdx.x & 31;
  if (wid >= N) return;
  const float4* row = (const float4*)(h + (size_t)wid * DFEAT);
  float4 p = row[lane];
  float4 q = row[lane + 32];
  float ss = p.x * p.x + p.y * p.y + p.z * p.z + p.w * p.w
           + q.x * q.x + q.y * q.y + q.z * q.z + q.w * q.w;
  ss = wave_sum32(ss);
  float r = rsqrtf(ss + AGNN_EPS);
  _Float16* orow = hn + (size_t)wid * DFEAT;
  v4h a, b;
  a[0] = (_Float16)(p.x * r); a[1] = (_Float16)(p.y * r);
  a[2] = (_Float16)(p.z * r); a[3] = (_Float16)(p.w * r);
  b[0] = (_Float16)(q.x * r); b[1] = (_Float16)(q.y * r);
  b[2] = (_Float16)(q.z * r); b[3] = (_Float16)(q.w * r);
  *(v4h*)(orow + 4 * lane)       = a;
  *(v4h*)(orow + 128 + 4 * lane) = b;
}

// ---------------- per-layer: edge cosine logits via WMMA ----------------------
// One wave handles 16 edges. A = 16 src rows (16x32 f16 tile per K-chunk),
// B = 16 dst rows as columns (32x16). Diagonal of D (16x16 f32) = edge dots.
// C/D layout: diag d<8 at {vgpr d, lane d}; d>=8 at {vgpr d-8, lane d+16}.

__global__ void k_edge_logits_wmma(const _Float16* __restrict__ hn,
                                   const int* __restrict__ src,
                                   const int* __restrict__ dst,
                                   const float* __restrict__ betas, int layer,
                                   float* __restrict__ logits, int E) {
  int wid  = (blockIdx.x * blockDim.x + threadIdx.x) >> 5;
  int lane = threadIdx.x & 31;
  int base = wid * 16;
  if (base >= E) return;                  // wave-uniform: EXEC stays all-ones
  const int i  = lane & 15;
  const int hi = lane >> 4;
  int ei = base + i;
  if (ei >= E) ei = E - 1;                // tail clamp (loads only)
  const _Float16* rowA = hn + (size_t)src[ei] * DFEAT;   // A: lane -> row M=i
  const _Float16* rowB = hn + (size_t)dst[ei] * DFEAT;   // B: lane -> col N=i
  v8f c = {};
#pragma unroll
  for (int k0 = 0; k0 < DFEAT; k0 += 32) {
    // A 16-bit 16x32: lanes 0-15 hold K = k0+0..15; lanes 16-31 K = k0+16..31
    v16h a = *(const v16h*)(rowA + k0 + hi * 16);
    // B 16-bit 32x16: lanes 0-15 hold K = k0+0..15; lanes 16-31 K = k0+16..31
    v16h b = *(const v16h*)(rowB + k0 + hi * 16);
    c = __builtin_amdgcn_wmma_f32_16x16x32_f16(false, a, false, b,
                                               (short)0, c, false, false);
  }
  float diag = 0.f;
#pragma unroll
  for (int r = 0; r < 8; ++r) diag = (lane == r || lane == r + 24) ? c[r] : diag;
  // compact diagonal into lanes 0..15 (edge order)
  float v = __shfl(diag, (lane < 8) ? lane : lane + 16, 32);
  float beta = betas[layer];
  if (lane < 16 && (base + lane) < E) logits[base + lane] = beta * v;
}

// ---------------- per-layer: softmax + weighted aggregation (atomic-free) -----

__global__ void k_aggregate(const float* __restrict__ hin,
                            const float* __restrict__ logits,
                            const int* __restrict__ csr_eid,
                            const int* __restrict__ src,
                            const int* __restrict__ offs,
                            float* __restrict__ hout, int N) {
  int n    = (blockIdx.x * blockDim.x + threadIdx.x) >> 5;
  int lane = threadIdx.x & 31;
  if (n >= N) return;
  const int start = offs[n], end = offs[n + 1];

  // segment max over this node's in-edges
  float m = -INFINITY;
  for (int s = start + lane; s < end; s += 32) m = fmaxf(m, logits[csr_eid[s]]);
  m = wave_max32(m);

  // segment sum of exp(l - m)
  float dsum = 0.f;
  for (int s = start + lane; s < end; s += 32) dsum += __expf(logits[csr_eid[s]] - m);
  dsum = wave_sum32(dsum);
  const float inv = 1.f / (dsum + AGNN_EPS);

  float4 acc0 = make_float4(0.f, 0.f, 0.f, 0.f);
  float4 acc1 = make_float4(0.f, 0.f, 0.f, 0.f);
  for (int cb = start; cb < end; cb += 32) {
    int cnt = end - cb; if (cnt > 32) cnt = 32;
    float alpha = 0.f; int srow = 0;
    if (lane < cnt) {
      int e  = csr_eid[cb + lane];
      alpha  = __expf(logits[e] - m) * inv;
      srow   = src[e];
      __builtin_prefetch(hin + (size_t)srow * DFEAT + 8 * lane, 0, 1);
    }
    for (int j = 0; j < cnt; ++j) {
      float aj = __shfl(alpha, j, 32);
      int   sj = __shfl(srow, j, 32);
      const float4* rp = (const float4*)(hin + (size_t)sj * DFEAT);
      float4 v0 = rp[lane];
      float4 v1 = rp[lane + 32];
      acc0.x = fmaf(aj, v0.x, acc0.x); acc0.y = fmaf(aj, v0.y, acc0.y);
      acc0.z = fmaf(aj, v0.z, acc0.z); acc0.w = fmaf(aj, v0.w, acc0.w);
      acc1.x = fmaf(aj, v1.x, acc1.x); acc1.y = fmaf(aj, v1.y, acc1.y);
      acc1.z = fmaf(aj, v1.z, acc1.z); acc1.w = fmaf(aj, v1.w, acc1.w);
    }
  }
  // fused ReLU + single coalesced row write
  float4 o0 = make_float4(fmaxf(acc0.x, 0.f), fmaxf(acc0.y, 0.f),
                          fmaxf(acc0.z, 0.f), fmaxf(acc0.w, 0.f));
  float4 o1 = make_float4(fmaxf(acc1.x, 0.f), fmaxf(acc1.y, 0.f),
                          fmaxf(acc1.z, 0.f), fmaxf(acc1.w, 0.f));
  float4* op = (float4*)(hout + (size_t)n * DFEAT);
  op[lane]      = o0;
  op[lane + 32] = o1;
}

// ---------------- host-side orchestration ------------------------------------

extern "C" void kernel_launch(void* const* d_in, const int* in_sizes, int n_in,
                              void* d_out, int out_size, void* d_ws, size_t ws_size,
                              hipStream_t stream) {
  (void)n_in; (void)out_size; (void)ws_size;
  const float* feats = (const float*)d_in[0];
  const int*   src   = (const int*)d_in[1];
  const int*   dst   = (const int*)d_in[2];
  const float* betas = (const float*)d_in[3];
  const int N = in_sizes[0] / DFEAT;
  const int E = in_sizes[1];
  const int L = in_sizes[3];
  float* out = (float*)d_out;

  uintptr_t p = (uintptr_t)d_ws;
  auto carve = [&](size_t bytes) -> void* {
    uintptr_t q = (p + 255) & ~(uintptr_t)255;
    p = q + bytes;
    return (void*)q;
  };
  float*    hbuf    = (float*)   carve((size_t)N * DFEAT * sizeof(float));
  _Float16* hn16    = (_Float16*)carve((size_t)N * DFEAT * sizeof(_Float16));
  float*    logits  = (float*)   carve((size_t)E * sizeof(float));
  int*      csr_eid = (int*)     carve((size_t)E * sizeof(int));
  int*      deg     = (int*)     carve((size_t)N * sizeof(int));
  int*      offs    = (int*)     carve((size_t)(N + 1) * sizeof(int));
  int*      cursor  = (int*)     carve((size_t)N * sizeof(int));

  // CSR build (graph is layer-invariant)
  hipMemsetAsync(deg, 0, (size_t)N * sizeof(int), stream);
  hipMemsetAsync(cursor, 0, (size_t)N * sizeof(int), stream);
  const int eb = (E + 255) / 256;
  k_count<<<eb, 256, 0, stream>>>(dst, deg, E);
  k_scan<<<1, 1024, 0, stream>>>(deg, offs, N);
  k_fill<<<eb, 256, 0, stream>>>(dst, offs, cursor, csr_eid, E);

  const int nodeb = (N + 7) / 8;                 // 8 waves/block, 1 node/wave
  const int dotw  = (E + 15) / 16;               // 16 edges/wave
  const int dotb  = (dotw + 7) / 8;

  const float* hin = feats;
  for (int l = 0; l < L; ++l) {
    // parity chosen so the final layer always lands in d_out (no aliasing)
    float* hout = (((L - 1 - l) & 1) == 0) ? out : hbuf;
    k_norm<<<nodeb, 256, 0, stream>>>(hin, hn16, N);
    k_edge_logits_wmma<<<dotb, 256, 0, stream>>>(hn16, src, dst, betas, l, logits, E);
    k_aggregate<<<nodeb, 256, 0, stream>>>(hin, logits, csr_eid, src, offs, hout, N);
    hin = hout;
  }
}
